// PatchDeepseekV3MoE_28037546508351
// MI455X (gfx1250) — compile-verified, matrix operand order checked by reading, not run on hardware
//
#include <hip/hip_runtime.h>
#include <hip/hip_bf16.h>

#define Tn   4096
#define Hdim 2048
#define Idim 1024
#define En   32
#define NSEG 34          // 32 routed + 2 shared pseudo-experts

typedef __attribute__((ext_vector_type(16))) __bf16 v16bf;
typedef __attribute__((ext_vector_type(8)))  __bf16 v8bf;
typedef __attribute__((ext_vector_type(8)))  float  v8f;

// ---------- fragment builders (ISA 7.12.2 16-bit layouts) ----------
// lane l: row/col = l%16 ; lanes<16 hold K {k..k+7, k+16..k+23}, lanes>=16 hold {k+8..k+15, k+24..k+31}
__device__ inline v16bf frag_g(const float* __restrict__ p0, const float* __restrict__ p1) {
    float4 a = *(const float4*)p0;
    float4 b = *(const float4*)(p0 + 4);
    float4 c = *(const float4*)p1;
    float4 d = *(const float4*)(p1 + 4);
    v16bf r;
    r[0]=(__bf16)a.x; r[1]=(__bf16)a.y; r[2]=(__bf16)a.z; r[3]=(__bf16)a.w;
    r[4]=(__bf16)b.x; r[5]=(__bf16)b.y; r[6]=(__bf16)b.z; r[7]=(__bf16)b.w;
    r[8]=(__bf16)c.x; r[9]=(__bf16)c.y; r[10]=(__bf16)c.z; r[11]=(__bf16)c.w;
    r[12]=(__bf16)d.x; r[13]=(__bf16)d.y; r[14]=(__bf16)d.z; r[15]=(__bf16)d.w;
    return r;
}

__device__ inline v16bf frag_l(const __bf16* p0) {   // from LDS bf16 (16B aligned)
    v8bf lo = *(const v8bf*)p0;
    v8bf hi = *(const v8bf*)(p0 + 16);
    v16bf r;
#pragma unroll
    for (int i = 0; i < 8; ++i) { r[i] = lo[i]; r[8 + i] = hi[i]; }
    return r;
}

__device__ inline v8f wmma_bf16(v16bf a, v16bf b, v8f c) {
    return __builtin_amdgcn_wmma_f32_16x16x32_bf16(false, a, false, b, (short)0, c, false, false);
}

// ---------- small kernels ----------
__global__ void k_zero4(float4* __restrict__ p, int n4) {
    int i = blockIdx.x * 256 + threadIdx.x;
    if (i < n4) p[i] = make_float4(0.f, 0.f, 0.f, 0.f);
}

__global__ void k_init(int* __restrict__ counts, int* __restrict__ fill) {
    int i = threadIdx.x;
    if (i < En) { counts[i] = 0; fill[i] = 0; }
}

// one wave per token: lane = expert
__global__ __launch_bounds__(256) void k_router(
    const float* __restrict__ x, const float* __restrict__ rw,
    const float* __restrict__ bias, int* __restrict__ counts,
    int* __restrict__ topk_idx, float* __restrict__ topk_w)
{
    __shared__ float sc_s[8][32];
    __shared__ float ss_s[8][32];
    int wave = threadIdx.x >> 5, lane = threadIdx.x & 31;
    int t = blockIdx.x * 8 + wave;

    const float* xr = x + (size_t)t * Hdim;
    const float* wr = rw + (size_t)lane * Hdim;
    float acc = 0.f;
    for (int h = 0; h < Hdim; h += 4) {
        float4 a = *(const float4*)(xr + h);
        float4 b = *(const float4*)(wr + h);
        acc += a.x * b.x + a.y * b.y + a.z * b.z + a.w * b.w;
    }
    float score = 1.f / (1.f + __expf(-acc));
    sc_s[wave][lane] = score + bias[lane];
    ss_s[wave][lane] = score;
    __syncthreads();

    if (lane == 0) {
        float sc[32], ss[32];
#pragma unroll
        for (int e = 0; e < 32; ++e) { sc[e] = sc_s[wave][e]; ss[e] = ss_s[wave][e]; }
        float gs[8];
#pragma unroll
        for (int g = 0; g < 8; ++g) {
            float m1 = -1e30f, m2 = -1e30f;
#pragma unroll
            for (int j = 0; j < 4; ++j) {
                float v = sc[g * 4 + j];
                if (v > m1) { m2 = m1; m1 = v; } else if (v > m2) { m2 = v; }
            }
            gs[g] = m1 + m2;
        }
        bool gmask[8] = {false,false,false,false,false,false,false,false};
        for (int s = 0; s < 4; ++s) {
            int bi = -1; float bv = -1e30f;
            for (int g = 0; g < 8; ++g)
                if (!gmask[g] && gs[g] > bv) { bv = gs[g]; bi = g; }
            gmask[bi] = true;
        }
        float scm[32];
#pragma unroll
        for (int e = 0; e < 32; ++e) scm[e] = gmask[e >> 2] ? sc[e] : 0.0f;
        int   idx[8]; float w[8]; float wsum = 0.f;
        for (int s = 0; s < 8; ++s) {
            int bi = 0; float bv = -1e30f;
            for (int e = 0; e < 32; ++e)
                if (scm[e] > bv) { bv = scm[e]; bi = e; }
            scm[bi] = -1e30f;
            idx[s] = bi; w[s] = ss[bi]; wsum += w[s];
        }
        float inv = 2.5f / (wsum + 1e-20f);
        for (int s = 0; s < 8; ++s) {
            topk_idx[t * 8 + s] = idx[s];
            topk_w[t * 8 + s]   = w[s] * inv;
            atomicAdd(&counts[idx[s]], 1);
        }
    }
}

__global__ void k_scan(const int* __restrict__ counts, int* __restrict__ seg) {
    if (threadIdx.x == 0 && blockIdx.x == 0) {
        int acc = 0;
        for (int e = 0; e < En; ++e) { seg[e] = acc; acc += counts[e]; }
        seg[32] = Tn * 8;
        seg[33] = Tn * 8 + Tn;
        seg[34] = Tn * 8 + 2 * Tn;
    }
}

__global__ void k_scatter(const int* __restrict__ topk_idx, const float* __restrict__ topk_w,
                          const int* __restrict__ seg, int* __restrict__ fill,
                          int* __restrict__ rows, float* __restrict__ wvals)
{
    int t = blockIdx.x * 256 + threadIdx.x;
    if (t >= Tn) return;
#pragma unroll
    for (int s = 0; s < 8; ++s) {
        int e = topk_idx[t * 8 + s];
        int pos = seg[e] + atomicAdd(&fill[e], 1);
        rows[pos] = t; wvals[pos] = topk_w[t * 8 + s];
    }
    rows[Tn * 8 + t]      = t; wvals[Tn * 8 + t]      = 1.0f;
    rows[Tn * 8 + Tn + t] = t; wvals[Tn * 8 + Tn + t] = 1.0f;
}

// ---------- fused expert kernel ----------
// wave layout: each wave owns ALL 64 rows (4 m-subtiles) x 32 cols (2 n-subtiles);
// every B (weight) fragment feeds 4 WMMAs.
__global__ __launch_bounds__(256) void k_moe(
    const float* __restrict__ x,
    const float* __restrict__ gate_w, const float* __restrict__ up_w,
    const float* __restrict__ down_w,
    const float* __restrict__ sh_g, const float* __restrict__ sh_u,
    const float* __restrict__ sh_d,
    const int* __restrict__ seg, const int* __restrict__ rows,
    const float* __restrict__ wvals, float* __restrict__ out)
{
    __shared__ __bf16 hbuf[64 * Idim];     // 128 KB
    __shared__ int    rows_s[64];
    __shared__ float  w_s[64];

    int e = blockIdx.y;
    int start = seg[e];
    int cnt = seg[e + 1] - start;
    int m0 = blockIdx.x * 64;
    if (m0 >= cnt) return;

    int tid = threadIdx.x;
    if (tid < 64) {
        int r = m0 + tid;
        if (r < cnt) { rows_s[tid] = rows[start + r]; w_s[tid] = wvals[start + r]; }
        else         { rows_s[tid] = rows[start + m0]; w_s[tid] = 0.0f; }
    }
    __syncthreads();

    const float *gp, *up, *dp; int ldd;
    if (e < En) {
        gp = gate_w + (size_t)e * Idim * Hdim;
        up = up_w   + (size_t)e * Idim * Hdim;
        dp = down_w + (size_t)e * Hdim * Idim;
        ldd = Idim;
    } else {
        int c = e - En;                       // SI=2048 split into two I=1024 chunks (exact)
        gp = sh_g + (size_t)c * Idim * Hdim;
        up = sh_u + (size_t)c * Idim * Hdim;
        dp = sh_d + (size_t)c * Idim;         // column offset within [H][2048]
        ldd = 2 * Idim;
    }

    int lane = tid & 31;
    int wn = tid >> 5;                 // 0..7 : 32-col slice within 256-col chunk
    int lhalf = lane >> 4, l16 = lane & 15;
    int klo = lhalf * 8;

    // x row pointers for the 4 m-subtiles (constant over k)
    const float* xr[4];
#pragma unroll
    for (int ms = 0; ms < 4; ++ms)
        xr[ms] = x + (size_t)rows_s[ms * 16 + l16] * Hdim;

    v8f z8;
#pragma unroll
    for (int i = 0; i < 8; ++i) z8[i] = 0.f;

    // ---- phase 1: h[64 x 1024] = silu(x g^T) * (x u^T) ----
    for (int c1 = 0; c1 < 4; ++c1) {
        int nb = c1 * 256 + wn * 32;
        v8f accG[4][2], accU[4][2];
#pragma unroll
        for (int ms = 0; ms < 4; ++ms)
#pragma unroll
            for (int ns = 0; ns < 2; ++ns) { accG[ms][ns] = z8; accU[ms][ns] = z8; }

#pragma unroll 2
        for (int k = 0; k < Hdim; k += 32) {
            v16bf a[4];
#pragma unroll
            for (int ms = 0; ms < 4; ++ms)
                a[ms] = frag_g(xr[ms] + k + klo, xr[ms] + k + 16 + klo);
#pragma unroll
            for (int ns = 0; ns < 2; ++ns) {
                const float* wb = gp + (size_t)(nb + ns * 16 + l16) * Hdim + k + klo;
                v16bf bg = frag_g(wb, wb + 16);
#pragma unroll
                for (int ms = 0; ms < 4; ++ms)
                    accG[ms][ns] = wmma_bf16(a[ms], bg, accG[ms][ns]);
                const float* ub = up + (size_t)(nb + ns * 16 + l16) * Hdim + k + klo;
                v16bf bu = frag_g(ub, ub + 16);
#pragma unroll
                for (int ms = 0; ms < 4; ++ms)
                    accU[ms][ns] = wmma_bf16(a[ms], bu, accU[ms][ns]);
            }
        }
#pragma unroll
        for (int ms = 0; ms < 4; ++ms)
#pragma unroll
            for (int ns = 0; ns < 2; ++ns)
#pragma unroll
                for (int v = 0; v < 8; ++v) {
                    float g = accG[ms][ns][v];
                    float u = accU[ms][ns][v];
                    float hval = (g / (1.0f + __expf(-g))) * u;
                    int row = ms * 16 + lhalf * 8 + v;   // C layout: VGPR v -> M=v / M=8+v
                    int col = nb + ns * 16 + l16;
                    hbuf[row * Idim + col] = (__bf16)hval;
                }
    }
    __syncthreads();

    // ---- phase 2: out_tile[64 x 2048] = (h @ d^T) * w, atomic add ----
    for (int nch = 0; nch < 8; ++nch) {
        int nb = nch * 256 + wn * 32;
        v8f acc[4][2];
#pragma unroll
        for (int ms = 0; ms < 4; ++ms)
#pragma unroll
            for (int ns = 0; ns < 2; ++ns) acc[ms][ns] = z8;

#pragma unroll 2
        for (int k = 0; k < Idim; k += 32) {
            v16bf a[4];
#pragma unroll
            for (int ms = 0; ms < 4; ++ms)
                a[ms] = frag_l(&hbuf[(ms * 16 + l16) * Idim + k + klo]);
#pragma unroll
            for (int ns = 0; ns < 2; ++ns) {
                const float* db = dp + (size_t)(nb + ns * 16 + l16) * ldd + k + klo;
                v16bf bd = frag_g(db, db + 16);
#pragma unroll
                for (int ms = 0; ms < 4; ++ms)
                    acc[ms][ns] = wmma_bf16(a[ms], bd, acc[ms][ns]);
            }
        }
#pragma unroll
        for (int ms = 0; ms < 4; ++ms) {
            int rbase = ms * 16 + lhalf * 8;
#pragma unroll
            for (int v = 0; v < 8; ++v) {
                int row = rbase + v;
                int tok = rows_s[row];
                float wgt = w_s[row];
#pragma unroll
                for (int ns = 0; ns < 2; ++ns) {
                    int col = nb + ns * 16 + l16;
                    unsafeAtomicAdd(&out[(size_t)tok * Hdim + col], acc[ms][ns][v] * wgt);
                }
            }
        }
    }
}

// ---------- host ----------
extern "C" void kernel_launch(void* const* d_in, const int* in_sizes, int n_in,
                              void* d_out, int out_size, void* d_ws, size_t ws_size,
                              hipStream_t stream) {
    const float* x    = (const float*)d_in[0];
    const float* rw   = (const float*)d_in[1];
    const float* bias = (const float*)d_in[2];
    const float* gw   = (const float*)d_in[3];
    const float* uw   = (const float*)d_in[4];
    const float* dw   = (const float*)d_in[5];
    const float* sg   = (const float*)d_in[6];
    const float* su   = (const float*)d_in[7];
    const float* sd   = (const float*)d_in[8];
    float* out = (float*)d_out;

    int* wsi = (int*)d_ws;
    int*   counts   = wsi + 0;
    int*   fill     = wsi + 32;
    int*   seg      = wsi + 64;
    int*   topk_idx = wsi + 128;
    float* topk_w   = (float*)(wsi + 128 + Tn * 8);
    int*   rows     = wsi + 128 + 2 * Tn * 8;
    float* wvals    = (float*)(wsi + 128 + 2 * Tn * 8 + (Tn * 8 + 2 * Tn));

    int n4 = Tn * Hdim / 4;
    k_zero4<<<(n4 + 255) / 256, 256, 0, stream>>>((float4*)out, n4);
    k_init<<<1, 64, 0, stream>>>(counts, fill);
    k_router<<<Tn / 8, 256, 0, stream>>>(x, rw, bias, counts, topk_idx, topk_w);
    k_scan<<<1, 1, 0, stream>>>(counts, seg);
    k_scatter<<<Tn / 256, 256, 0, stream>>>(topk_idx, topk_w, seg, fill, rows, wvals);
    dim3 grid(64, NSEG);
    k_moe<<<grid, 256, 0, stream>>>(x, gw, uw, dw, sg, su, sd, seg, rows, wvals, out);
}
